// MixtureOfExperts_78073915506798
// MI455X (gfx1250) — matrix operand order repaired and checked by reading, not compiled, not run on hardware
//
#include <hip/hip_runtime.h>

typedef __attribute__((ext_vector_type(16))) _Float16     v16h;
typedef __attribute__((ext_vector_type(8)))  float        v8f;
typedef __attribute__((ext_vector_type(4)))  unsigned int v4u;

#define B_TOT   524288
#define IN_DIM  59
#define NE      4
#define HID     64
#define EMBD    32
#define GHID    32
#define EPSBN   1e-5f

union HFrag { v16h h; v4u u[2]; };

__device__ __forceinline__ v8f wmma_f16(v16h a, v16h b, v8f c) {
    // D = A(16x32 f16) * B(32x16 f16) + C(16x16 f32)
    return __builtin_amdgcn_wmma_f32_16x16x32_f16(
        /*neg_a=*/false, a, /*neg_b=*/false, b,
        /*c_mod=*/(short)0, c, /*reuse_a=*/false, /*reuse_b=*/false);
}

// A fragment (16x32 f16) from row-major LDS tile with row stride 64 halves.
// ISA 7.12.2: lanes 0-15 hold M=lane, K = {0..7, 16..23}; lanes 16-31 hold
// M=lane-16, K = {8..15, 24..31}  (plus kbase offset for the K=32..63 chunk).
__device__ __forceinline__ v16h load_a_lds(const _Float16* base, int mt, int kbase, int lane) {
    int row = mt * 16 + (lane & 15);
    int klo = kbase + ((lane >> 4) << 3);
    const _Float16* p = base + row * 64 + klo;
    HFrag f;
    f.u[0] = *(const v4u*)(p);        // K = klo .. klo+7
    f.u[1] = *(const v4u*)(p + 16);   // K = klo+16 .. klo+23
    return f.h;
}

// B fragment (32x16 f16) from transposed (NxK, K-stride 64) f16 weights in global.
// lanes 0-15: N=lane, K = kbase+{0..7,16..23} ; lanes 16-31: N=lane-16, K = kbase+{8..15,24..31}.
__device__ __forceinline__ v16h load_b_glb(const _Float16* wt, int nt, int kbase, int lane) {
    int n  = nt * 16 + (lane & 15);
    int kk = kbase + ((lane >> 4) << 3);
    const _Float16* p = wt + n * 64 + kk;
    HFrag f;
    f.u[0] = *(const v4u*)(p);        // K = kk .. kk+7
    f.u[1] = *(const v4u*)(p + 16);   // K = kk+16 .. kk+23
    return f.h;
}

// ---------------------------------------------------------------------------
// Prep: transpose/pad weights to f16 (N x 64 layout) and fold BN into affine.
// ---------------------------------------------------------------------------
__global__ void moe_prep(const float* __restrict__ w1, const float* __restrict__ b1,
                         const float* __restrict__ bn1g, const float* __restrict__ bn1b,
                         const float* __restrict__ bn1m, const float* __restrict__ bn1v,
                         const float* __restrict__ w2, const float* __restrict__ b2,
                         const float* __restrict__ bn2g, const float* __restrict__ bn2b,
                         const float* __restrict__ bn2m, const float* __restrict__ bn2v,
                         const float* __restrict__ w3, const float* __restrict__ gw1,
                         _Float16* __restrict__ W1T, _Float16* __restrict__ W2T,
                         _Float16* __restrict__ W3T, _Float16* __restrict__ GW1T,
                         float* __restrict__ al1, float* __restrict__ be1,
                         float* __restrict__ al2, float* __restrict__ be2) {
    const int stride = blockDim.x * gridDim.x;
    const int t0 = blockIdx.x * blockDim.x + threadIdx.x;

    // W1T[e][n][k] = (k<59) ? w1[e][k][n] : 0      (w1: (E,59,64))
    for (int i = t0; i < NE * 64 * 64; i += stride) {
        int e = i >> 12, n = (i >> 6) & 63, k = i & 63;
        float v = (k < IN_DIM) ? w1[((size_t)e * IN_DIM + k) * HID + n] : 0.0f;
        W1T[i] = (_Float16)v;
    }
    // W2T[e][n][k] = w2[e][k][n]                   (w2: (E,64,64))
    for (int i = t0; i < NE * 64 * 64; i += stride) {
        int e = i >> 12, n = (i >> 6) & 63, k = i & 63;
        W2T[i] = (_Float16)w2[((size_t)e * HID + k) * HID + n];
    }
    // W3T[e][n][k] = w3[e][k][n]                   (w3: (E,64,32)), n<32
    for (int i = t0; i < NE * EMBD * 64; i += stride) {
        int e = i >> 11, n = (i >> 6) & 31, k = i & 63;
        W3T[i] = (_Float16)w3[((size_t)e * HID + k) * EMBD + n];
    }
    // GW1T[n][k] = (k<59) ? gw1[k][n] : 0          (gw1: (59,32))
    for (int i = t0; i < GHID * 64; i += stride) {
        int n = i >> 6, k = i & 63;
        float v = (k < IN_DIM) ? gw1[(size_t)k * GHID + n] : 0.0f;
        GW1T[i] = (_Float16)v;
    }
    // folded BN affines: y = acc*alpha + beta, beta absorbs linear bias too
    for (int i = t0; i < NE * HID; i += stride) {
        float a1 = bn1g[i] * rsqrtf(bn1v[i] + EPSBN);
        al1[i] = a1;
        be1[i] = (b1[i] - bn1m[i]) * a1 + bn1b[i];
        float a2 = bn2g[i] * rsqrtf(bn2v[i] + EPSBN);
        al2[i] = a2;
        be2[i] = (b2[i] - bn2m[i]) * a2 + bn2b[i];
    }
}

// ---------------------------------------------------------------------------
// Main: 64 rows/block, 4 waves, wave w == expert w.
// B fragments are hoisted into registers once per layer (mt-invariant).
// ---------------------------------------------------------------------------
__global__ __launch_bounds__(128) void moe_main(
    const float* __restrict__ x,
    const _Float16* __restrict__ W1T, const _Float16* __restrict__ W2T,
    const _Float16* __restrict__ W3T, const _Float16* __restrict__ GW1T,
    const float* __restrict__ al1, const float* __restrict__ be1,
    const float* __restrict__ al2, const float* __restrict__ be2,
    const float* __restrict__ b3, const float* __restrict__ wp,
    const float* __restrict__ bp, const float* __restrict__ gb1,
    const float* __restrict__ gw2, const float* __restrict__ gb2,
    float* __restrict__ out) {

    __shared__ __align__(16) _Float16 sX[64 * 64];        //  8 KB  x tile (f16, padded)
    __shared__ __align__(16) _Float16 sH[NE][64 * 64];    // 32 KB  per-expert hidden
    __shared__ float sGH[64][GHID];                       //  8 KB  gate hidden
    __shared__ float sPred[NE][64];                       //  1 KB  expert predictions

    const int tid  = threadIdx.x;
    const int lane = tid & 31;
    const int wave = tid >> 5;           // expert id
    const int r0   = blockIdx.x * 64;

    // ---- stage A: load x tile -> f16 LDS, zero-pad columns 59..63 ----
    for (int i = tid; i < 64 * 64; i += 128) {
        int r = i >> 6, c = i & 63;
        float v = (c < IN_DIM) ? x[(size_t)(r0 + r) * IN_DIM + c] : 0.0f;
        sX[i] = (_Float16)v;
    }
    __syncthreads();

    // ---- stage B: gate hidden = relu(x@gw1 + gb1) via WMMA ----
    // wave w computes mt=w, nt=0..1 (disjoint rows, covers 64x32).
    {
        v16h gb[4];
        #pragma unroll
        for (int nt = 0; nt < 2; ++nt) {
            gb[2 * nt]     = load_b_glb(GW1T, nt, 0, lane);
            gb[2 * nt + 1] = load_b_glb(GW1T, nt, 32, lane);
        }
        const int mt = wave;
        v16h a0 = load_a_lds(sX, mt, 0, lane);
        v16h a1 = load_a_lds(sX, mt, 32, lane);
        #pragma unroll
        for (int nt = 0; nt < 2; ++nt) {
            v8f acc = {};
            acc = wmma_f16(a0, gb[2 * nt], acc);
            acc = wmma_f16(a1, gb[2 * nt + 1], acc);
            int n = nt * 16 + (lane & 15);
            float bb = gb1[n];
            #pragma unroll
            for (int g = 0; g < 8; ++g) {
                int m = mt * 16 + g + ((lane >> 4) << 3);
                float v = acc[g] + bb;
                sGH[m][n] = v > 0.0f ? v : 0.0f;
            }
        }
    }

    const int e = wave;
    _Float16* sHe = sH[e];

    // ---- layer 1: h = relu(BN(x @ w1 + b1))  -> sH[e] ----
    {
        const _Float16* w1e = W1T + e * 64 * 64;
        v16h bf[8];
        float al[4], be[4];
        #pragma unroll
        for (int nt = 0; nt < 4; ++nt) {
            bf[2 * nt]     = load_b_glb(w1e, nt, 0, lane);
            bf[2 * nt + 1] = load_b_glb(w1e, nt, 32, lane);
            al[nt] = al1[e * HID + nt * 16 + (lane & 15)];
            be[nt] = be1[e * HID + nt * 16 + (lane & 15)];
        }
        for (int mt = 0; mt < 4; ++mt) {
            v16h a0 = load_a_lds(sX, mt, 0, lane);
            v16h a1 = load_a_lds(sX, mt, 32, lane);
            #pragma unroll
            for (int nt = 0; nt < 4; ++nt) {
                v8f acc = {};
                acc = wmma_f16(a0, bf[2 * nt], acc);
                acc = wmma_f16(a1, bf[2 * nt + 1], acc);
                int n = nt * 16 + (lane & 15);
                #pragma unroll
                for (int g = 0; g < 8; ++g) {
                    int m = mt * 16 + g + ((lane >> 4) << 3);
                    float v = acc[g] * al[nt] + be[nt];
                    sHe[m * 64 + n] = (_Float16)(v > 0.0f ? v : 0.0f);
                }
            }
        }
    }

    // ---- layer 2: h = relu(BN(h @ w2 + b2))  in-place in sH[e] ----
    // (A fragments for row-tile mt are fully loaded before any store to those rows)
    {
        const _Float16* w2e = W2T + e * 64 * 64;
        v16h bf[8];
        float al[4], be[4];
        #pragma unroll
        for (int nt = 0; nt < 4; ++nt) {
            bf[2 * nt]     = load_b_glb(w2e, nt, 0, lane);
            bf[2 * nt + 1] = load_b_glb(w2e, nt, 32, lane);
            al[nt] = al2[e * HID + nt * 16 + (lane & 15)];
            be[nt] = be2[e * HID + nt * 16 + (lane & 15)];
        }
        for (int mt = 0; mt < 4; ++mt) {
            v16h a0 = load_a_lds(sHe, mt, 0, lane);
            v16h a1 = load_a_lds(sHe, mt, 32, lane);
            #pragma unroll
            for (int nt = 0; nt < 4; ++nt) {
                v8f acc = {};
                acc = wmma_f16(a0, bf[2 * nt], acc);
                acc = wmma_f16(a1, bf[2 * nt + 1], acc);
                int n = nt * 16 + (lane & 15);
                #pragma unroll
                for (int g = 0; g < 8; ++g) {
                    int m = mt * 16 + g + ((lane >> 4) << 3);
                    float v = acc[g] * al[nt] + be[nt];
                    sHe[m * 64 + n] = (_Float16)(v > 0.0f ? v : 0.0f);
                }
            }
        }
    }

    // ---- layer 3 + predictor: preds = (h @ w3 + b3) . wp + bp ----
    {
        const _Float16* w3e = W3T + e * EMBD * 64;
        const float bpv = bp[e];
        v16h bf[4];
        float b3v[2], wpv[2];
        #pragma unroll
        for (int nt = 0; nt < 2; ++nt) {
            bf[2 * nt]     = load_b_glb(w3e, nt, 0, lane);
            bf[2 * nt + 1] = load_b_glb(w3e, nt, 32, lane);
            b3v[nt] = b3[e * EMBD + nt * 16 + (lane & 15)];
            wpv[nt] = wp[e * EMBD + nt * 16 + (lane & 15)];
        }
        for (int mt = 0; mt < 4; ++mt) {
            v16h a0 = load_a_lds(sHe, mt, 0, lane);
            v16h a1 = load_a_lds(sHe, mt, 32, lane);
            float part[8];
            #pragma unroll
            for (int g = 0; g < 8; ++g) part[g] = 0.0f;
            #pragma unroll
            for (int nt = 0; nt < 2; ++nt) {
                v8f acc = {};
                acc = wmma_f16(a0, bf[2 * nt], acc);
                acc = wmma_f16(a1, bf[2 * nt + 1], acc);
                #pragma unroll
                for (int g = 0; g < 8; ++g)
                    part[g] += (acc[g] + b3v[nt]) * wpv[nt];
            }
            // reduce over the 16 lanes of each half-wave (n = 0..31 folded in)
            #pragma unroll
            for (int g = 0; g < 8; ++g) {
                float s = part[g];
                s += __shfl_xor(s, 1);
                s += __shfl_xor(s, 2);
                s += __shfl_xor(s, 4);
                s += __shfl_xor(s, 8);
                if ((lane & 15) == 0) {
                    int m = mt * 16 + g + ((lane >> 4) << 3);
                    sPred[e][m] = s + bpv;
                }
            }
        }
    }
    __syncthreads();

    // ---- final: gate layer 2 + softmax + mixture, one thread per row ----
    if (tid < 64) {
        float g2[NE];
        #pragma unroll
        for (int q = 0; q < NE; ++q) g2[q] = gb2[q];
        for (int j = 0; j < GHID; ++j) {
            float h = sGH[tid][j];
            #pragma unroll
            for (int q = 0; q < NE; ++q) g2[q] += h * gw2[j * NE + q];
        }
        float mx = fmaxf(fmaxf(g2[0], g2[1]), fmaxf(g2[2], g2[3]));
        float pg[NE], s = 0.0f;
        #pragma unroll
        for (int q = 0; q < NE; ++q) { pg[q] = __expf(g2[q] - mx); s += pg[q]; }
        float inv = 1.0f / s;
        float res = 0.0f;
        #pragma unroll
        for (int q = 0; q < NE; ++q) res += sPred[q][tid] * pg[q];
        out[r0 + tid] = res * inv;
    }
}

extern "C" void kernel_launch(void* const* d_in, const int* in_sizes, int n_in,
                              void* d_out, int out_size, void* d_ws, size_t ws_size,
                              hipStream_t stream) {
    const float* x     = (const float*)d_in[0];
    const float* w1    = (const float*)d_in[1];
    const float* b1    = (const float*)d_in[2];
    const float* bn1g  = (const float*)d_in[3];
    const float* bn1b  = (const float*)d_in[4];
    const float* bn1m  = (const float*)d_in[5];
    const float* bn1v  = (const float*)d_in[6];
    const float* w2    = (const float*)d_in[7];
    const float* b2    = (const float*)d_in[8];
    const float* bn2g  = (const float*)d_in[9];
    const float* bn2b  = (const float*)d_in[10];
    const float* bn2m  = (const float*)d_in[11];
    const float* bn2v  = (const float*)d_in[12];
    const float* w3    = (const float*)d_in[13];
    const float* b3    = (const float*)d_in[14];
    const float* wp    = (const float*)d_in[15];
    const float* bp    = (const float*)d_in[16];
    const float* gw1   = (const float*)d_in[17];
    const float* gb1   = (const float*)d_in[18];
    const float* gw2   = (const float*)d_in[19];
    const float* gb2   = (const float*)d_in[20];
    float* out = (float*)d_out;

    char* ws = (char*)d_ws;
    _Float16* W1T  = (_Float16*)(ws + 0);        // 32768 B
    _Float16* W2T  = (_Float16*)(ws + 32768);    // 32768 B
    _Float16* W3T  = (_Float16*)(ws + 65536);    // 16384 B
    _Float16* GW1T = (_Float16*)(ws + 81920);    //  4096 B
    float*    al1  = (float*)(ws + 86016);
    float*    be1  = (float*)(ws + 87040);
    float*    al2  = (float*)(ws + 88064);
    float*    be2  = (float*)(ws + 89088);       // ends 90112 B

    moe_prep<<<64, 256, 0, stream>>>(w1, b1, bn1g, bn1b, bn1m, bn1v,
                                     w2, b2, bn2g, bn2b, bn2m, bn2v,
                                     w3, gw1, W1T, W2T, W3T, GW1T,
                                     al1, be1, al2, be2);

    moe_main<<<B_TOT / 64, 128, 0, stream>>>(x, W1T, W2T, W3T, GW1T,
                                             al1, be1, al2, be2,
                                             b3, wp, bp, gb1, gw2, gb2, out);
}
